// MambaBlock_66374424592996
// MI455X (gfx1250) — matrix operand order repaired and checked by reading, not compiled, not run on hardware
//
#include <hip/hip_runtime.h>

// ---------------- types ----------------
typedef __bf16 bf16_t;
typedef __attribute__((ext_vector_type(16))) __bf16 v16bf;
typedef __attribute__((ext_vector_type(8)))  __bf16 v8bf;
typedef __attribute__((ext_vector_type(8)))  float  v8f;

// explicit global (addrspace 1) pointers -> force global_load/global_store
typedef __attribute__((address_space(1))) const __bf16* gbf_p;
typedef __attribute__((address_space(1))) const v8bf*   gv8bf_p;
typedef __attribute__((address_space(1))) float*        gf32_p;

// ---------------- problem constants ----------------
#define LSEQ 2048
#define DMODEL 768
#define DINNER 1536
#define NSTATE 16
#define KCONV 4
#define RRANK 48
#define RPAD 64          // dt_rank padded to WMMA K granularity
#define NXPROJ 80        // R + 2N (logical)
#define NXPAD 128        // padded to N-tile granularity -> branch-free GEMM

__device__ __forceinline__ bf16_t f2bf(float f) {
    unsigned u = __builtin_bit_cast(unsigned, f);
    unsigned r = (u + 0x7FFFu + ((u >> 16) & 1u)) >> 16;
    unsigned short s = (unsigned short)r;
    return __builtin_bit_cast(bf16_t, s);
}

__device__ __forceinline__ v16bf load_frag16_g(gbf_p p0, gbf_p p1) {
    v8bf lo = *(gv8bf_p)p0;
    v8bf hi = *(gv8bf_p)p1;
    return __builtin_shufflevector(lo, hi,
        0, 1, 2, 3, 4, 5, 6, 7, 8, 9, 10, 11, 12, 13, 14, 15);
}

// ---------------- fp32 -> bf16 convert ----------------
__global__ __launch_bounds__(256) void f32_to_bf16_kernel(const float* __restrict__ in,
                                                          bf16_t* __restrict__ out, int n) {
    int i = blockIdx.x * 256 + threadIdx.x;
    if (i < n) out[i] = f2bf(in[i]);
}

// w_dt [DINNER, RRANK] -> bf16 [DINNER, RPAD] zero padded
__global__ __launch_bounds__(256) void pack_wdt_kernel(const float* __restrict__ w_dt,
                                                       bf16_t* __restrict__ out) {
    int i = blockIdx.x * 256 + threadIdx.x;
    if (i >= DINNER * RPAD) return;
    int row = i >> 6, c = i & 63;
    out[i] = f2bf(c < RRANK ? w_dt[row * RRANK + c] : 0.f);
}

// w_xproj [NXPROJ, DINNER] -> bf16 [NXPAD, DINNER] zero padded rows
__global__ __launch_bounds__(256) void pack_wxproj_kernel(const float* __restrict__ w_xproj,
                                                          bf16_t* __restrict__ out) {
    int i = blockIdx.x * 256 + threadIdx.x;
    if (i >= NXPAD * DINNER) return;
    int row = i / DINNER, c = i % DINNER;
    out[i] = f2bf(row < NXPROJ ? w_xproj[(size_t)row * DINNER + c] : 0.f);
}

// x_dbl[:, :RRANK] (stride NXPAD) -> bf16 [LSEQ, RPAD] zero padded
__global__ __launch_bounds__(256) void pack_dtin_kernel(const float* __restrict__ x_dbl,
                                                        bf16_t* __restrict__ out) {
    int i = blockIdx.x * 256 + threadIdx.x;
    if (i >= LSEQ * RPAD) return;
    int l = i >> 6, c = i & 63;
    out[i] = f2bf(c < RRANK ? x_dbl[(size_t)l * NXPAD + c] : 0.f);
}

// ---------------- WMMA GEMM: C[M,N] = A[M,K] x Wt[N,K]^T ----------------
// A, Wt bf16 row-major; C fp32 row-major. M % 128 == 0, N % 64 == 0, K % 64 == 0.
// One wave computes a 32x64 tile (2 M-subtiles x 4 N-subtiles).
// K-loop unrolled by 64 with ping-pong fragment buffers: no register copies,
// loads for one buffer overlap WMMAs on the other. Branch-free inner loop.
#define GEMM_WAVES 4
#define GEMM_MT 2
__global__ __launch_bounds__(GEMM_WAVES * 32)
void gemm_bf16_wmma(const bf16_t* __restrict__ A, const bf16_t* __restrict__ Wt,
                    float* __restrict__ C, int M, int N, int K) {
    const int lane = threadIdx.x & 31;
    const int wave = threadIdx.x >> 5;
    const int m0 = (blockIdx.y * GEMM_WAVES + wave) * (16 * GEMM_MT);
    const int n0 = blockIdx.x * 64;

    const int row = lane & 15;       // M-row (A) / N-col (B) within 16-tile
    const int kh  = lane >> 4;       // K-half selector

    // A: lane holds row m, K = kh*8..+7 and 16+kh*8..+7 (16-bit A 16x32 layout)
    gbf_p ap[GEMM_MT];
#pragma unroll
    for (int mt = 0; mt < GEMM_MT; ++mt)
        ap[mt] = (gbf_p)(unsigned long long)(A + (size_t)(m0 + mt * 16 + row) * K + kh * 8);
    // B: lane holds column n (= row of Wt), K = kh*8..+7 and 16+kh*8..+7 (same as A)
    gbf_p bp[4];
#pragma unroll
    for (int nt = 0; nt < 4; ++nt)
        bp[nt] = (gbf_p)(unsigned long long)(Wt + (size_t)(n0 + nt * 16 + row) * K + kh * 8);

    v8f acc[GEMM_MT][4] = {};
    v16bf a0[GEMM_MT], a1[GEMM_MT], b0[4], b1[4];

    // prologue: buffer0 <- k=0, buffer1 <- k=32
#pragma unroll
    for (int mt = 0; mt < GEMM_MT; ++mt) {
        a0[mt] = load_frag16_g(ap[mt],      ap[mt] + 16);
        a1[mt] = load_frag16_g(ap[mt] + 32, ap[mt] + 48);
    }
#pragma unroll
    for (int nt = 0; nt < 4; ++nt) {
        b0[nt] = load_frag16_g(bp[nt],      bp[nt] + 16);
        b1[nt] = load_frag16_g(bp[nt] + 32, bp[nt] + 48);
    }

    for (int k0 = 64; k0 < K; k0 += 64) {
#pragma unroll
        for (int mt = 0; mt < GEMM_MT; ++mt) ap[mt] += 64;
#pragma unroll
        for (int nt = 0; nt < 4; ++nt)       bp[nt] += 64;

        // consume buffer0, then refill it from k0
#pragma unroll
        for (int mt = 0; mt < GEMM_MT; ++mt)
#pragma unroll
            for (int nt = 0; nt < 4; ++nt)
                acc[mt][nt] = __builtin_amdgcn_wmma_f32_16x16x32_bf16(
                    false, a0[mt], false, b0[nt], (short)0, acc[mt][nt], false, false);
#pragma unroll
        for (int mt = 0; mt < GEMM_MT; ++mt) a0[mt] = load_frag16_g(ap[mt], ap[mt] + 16);
#pragma unroll
        for (int nt = 0; nt < 4; ++nt)       b0[nt] = load_frag16_g(bp[nt], bp[nt] + 16);

        // consume buffer1, then refill it from k0+32
#pragma unroll
        for (int mt = 0; mt < GEMM_MT; ++mt)
#pragma unroll
            for (int nt = 0; nt < 4; ++nt)
                acc[mt][nt] = __builtin_amdgcn_wmma_f32_16x16x32_bf16(
                    false, a1[mt], false, b1[nt], (short)0, acc[mt][nt], false, false);
#pragma unroll
        for (int mt = 0; mt < GEMM_MT; ++mt) a1[mt] = load_frag16_g(ap[mt] + 32, ap[mt] + 48);
#pragma unroll
        for (int nt = 0; nt < 4; ++nt)       b1[nt] = load_frag16_g(bp[nt] + 32, bp[nt] + 48);
    }

    // epilogue: consume both buffers
#pragma unroll
    for (int mt = 0; mt < GEMM_MT; ++mt)
#pragma unroll
        for (int nt = 0; nt < 4; ++nt)
            acc[mt][nt] = __builtin_amdgcn_wmma_f32_16x16x32_bf16(
                false, a0[mt], false, b0[nt], (short)0, acc[mt][nt], false, false);
#pragma unroll
    for (int mt = 0; mt < GEMM_MT; ++mt)
#pragma unroll
        for (int nt = 0; nt < 4; ++nt)
            acc[mt][nt] = __builtin_amdgcn_wmma_f32_16x16x32_bf16(
                false, a1[mt], false, b1[nt], (short)0, acc[mt][nt], false, false);

    // C/D layout: lane l, vreg r -> (mBase + (l>>4)*8 + r, nb + (l&15))
#pragma unroll
    for (int mt = 0; mt < GEMM_MT; ++mt) {
#pragma unroll
        for (int nt = 0; nt < 4; ++nt) {
            gf32_p cRow = (gf32_p)(unsigned long long)
                (C + (size_t)(m0 + mt * 16 + kh * 8) * N + n0 + nt * 16 + row);
#pragma unroll
            for (int r = 0; r < 8; ++r) cRow[(size_t)r * N] = acc[mt][nt][r];
        }
    }
}

// ---------------- causal depthwise conv1d + SiLU ----------------
__global__ __launch_bounds__(256) void conv_silu_kernel(const float* __restrict__ xz,
                                                        const float* __restrict__ w_conv,
                                                        const float* __restrict__ b_conv,
                                                        float* __restrict__ xc,
                                                        bf16_t* __restrict__ xcbf) {
    int idx = blockIdx.x * 256 + threadIdx.x;
    if (idx >= LSEQ * DINNER) return;
    int l = idx / DINNER, c = idx % DINNER;
    float acc = b_conv[c];
#pragma unroll
    for (int j = 0; j < KCONV; ++j) {
        int ll = l - (KCONV - 1) + j;
        if (ll >= 0) acc += w_conv[c * KCONV + j] * xz[(size_t)ll * (2 * DINNER) + c];
    }
    float s = acc / (1.f + __expf(-acc));   // SiLU
    xc[idx] = s;
    xcbf[idx] = f2bf(s);
}

// ---------------- selective scan (sequential over L) ----------------
#define SCAN_TPB 128
#define SCAN_CHUNK 64
__global__ __launch_bounds__(SCAN_TPB)
void scan_kernel(const float* __restrict__ dt_raw,   // [L, DINNER]
                 const float* __restrict__ b_dt,     // [DINNER]
                 const float* __restrict__ x_dbl,    // [L, NXPAD] (B at 48, C at 64)
                 const float* __restrict__ xc,       // [L, DINNER]
                 const float* __restrict__ xz,       // [L, 2*DINNER] (z in upper half)
                 const float* __restrict__ A_log,    // [DINNER, NSTATE]
                 const float* __restrict__ D_param,  // [DINNER]
                 bf16_t* __restrict__ ybf) {         // [L, DINNER]
    __shared__ float sB[SCAN_CHUNK][NSTATE];
    __shared__ float sC[SCAN_CHUNK][NSTATE];
    const int d = blockIdx.x * SCAN_TPB + threadIdx.x;

    float A[NSTATE], h[NSTATE];
    const float bdt = b_dt[d];
    const float Dp = D_param[d];
#pragma unroll
    for (int n = 0; n < NSTATE; ++n) {
        A[n] = -__expf(A_log[(size_t)d * NSTATE + n]);
        h[n] = 0.f;
    }

    for (int lc = 0; lc < LSEQ; lc += SCAN_CHUNK) {
        __syncthreads();
        for (int t = threadIdx.x; t < SCAN_CHUNK * 32; t += SCAN_TPB) {
            int rr = t >> 5, cc = t & 31;
            float v = x_dbl[(size_t)(lc + rr) * NXPAD + RRANK + cc];
            if (cc < NSTATE) sB[rr][cc] = v; else sC[rr][cc - NSTATE] = v;
        }
        __syncthreads();
        for (int i = 0; i < SCAN_CHUNK; ++i) {
            const int l = lc + i;
            float dtv = dt_raw[(size_t)l * DINNER + d] + bdt;
            dtv = (dtv > 20.f) ? dtv : __logf(1.f + __expf(dtv));   // softplus
            const float u = xc[(size_t)l * DINNER + d];
            const float du = dtv * u;
            float y = 0.f;
#pragma unroll
            for (int n = 0; n < NSTATE; ++n) {
                float dA = __expf(dtv * A[n]);
                h[n] = dA * h[n] + du * sB[i][n];
                y += h[n] * sC[i][n];
            }
            y += Dp * u;
            const float z = xz[(size_t)l * (2 * DINNER) + DINNER + d];
            y *= z / (1.f + __expf(-z));    // y * silu(z)
            ybf[(size_t)l * DINNER + d] = f2bf(y);
        }
    }
}

// ---------------- residual + RMSNorm ----------------
__global__ __launch_bounds__(256) void rmsnorm_kernel(const float* __restrict__ out_lin,
                                                      const float* __restrict__ x,
                                                      const float* __restrict__ w_norm,
                                                      float* __restrict__ out) {
    const int l = blockIdx.x;
    __shared__ float red[8];
    float vals[3];
    float ss = 0.f;
#pragma unroll
    for (int i = 0; i < 3; ++i) {
        int c = threadIdx.x + i * 256;
        float r = out_lin[(size_t)l * DMODEL + c] + x[(size_t)l * DMODEL + c];
        vals[i] = r;
        ss += r * r;
    }
#pragma unroll
    for (int off = 16; off > 0; off >>= 1) ss += __shfl_down(ss, off, 32);
    if ((threadIdx.x & 31) == 0) red[threadIdx.x >> 5] = ss;
    __syncthreads();
    if (threadIdx.x < 8) {
        float v = red[threadIdx.x];
        v += __shfl_down(v, 4, 32);
        v += __shfl_down(v, 2, 32);
        v += __shfl_down(v, 1, 32);
        if (threadIdx.x == 0) red[0] = v;
    }
    __syncthreads();
    const float scale = __frsqrt_rn(red[0] / (float)DMODEL + 1e-5f);
#pragma unroll
    for (int i = 0; i < 3; ++i) {
        int c = threadIdx.x + i * 256;
        out[(size_t)l * DMODEL + c] = vals[i] * scale * w_norm[c];
    }
}

// ---------------- host orchestration ----------------
extern "C" void kernel_launch(void* const* d_in, const int* in_sizes, int n_in,
                              void* d_out, int out_size, void* d_ws, size_t ws_size,
                              hipStream_t stream) {
    const float* x       = (const float*)d_in[0];
    const float* w_in    = (const float*)d_in[1];
    const float* w_conv  = (const float*)d_in[2];
    const float* b_conv  = (const float*)d_in[3];
    const float* w_xproj = (const float*)d_in[4];
    const float* w_dt    = (const float*)d_in[5];
    const float* b_dt    = (const float*)d_in[6];
    const float* A_log   = (const float*)d_in[7];
    const float* D_param = (const float*)d_in[8];
    const float* w_out   = (const float*)d_in[9];
    const float* w_norm  = (const float*)d_in[10];
    (void)in_sizes; (void)n_in; (void)out_size; (void)ws_size;

    char* ws = (char*)d_ws;
    size_t off = 0;
    auto carve = [&](size_t bytes) -> char* {
        char* p = ws + off;
        off = (off + bytes + 255) & ~(size_t)255;
        return p;
    };

    bf16_t* xbf     = (bf16_t*)carve((size_t)LSEQ * DMODEL * 2);
    bf16_t* winbf   = (bf16_t*)carve((size_t)2 * DINNER * DMODEL * 2);
    bf16_t* wxbf    = (bf16_t*)carve((size_t)NXPAD * DINNER * 2);
    bf16_t* wdtbf   = (bf16_t*)carve((size_t)DINNER * RPAD * 2);
    bf16_t* woutbf  = (bf16_t*)carve((size_t)DMODEL * DINNER * 2);
    float*  xz      = (float*)carve((size_t)LSEQ * 2 * DINNER * 4);
    float*  xc      = (float*)carve((size_t)LSEQ * DINNER * 4);
    bf16_t* xcbf    = (bf16_t*)carve((size_t)LSEQ * DINNER * 2);
    float*  x_dbl   = (float*)carve((size_t)LSEQ * NXPAD * 4);
    bf16_t* dtinbf  = (bf16_t*)carve((size_t)LSEQ * RPAD * 2);
    float*  dt_raw  = (float*)carve((size_t)LSEQ * DINNER * 4);
    bf16_t* ybf     = (bf16_t*)carve((size_t)LSEQ * DINNER * 2);
    float*  outlin  = (float*)carve((size_t)LSEQ * DMODEL * 4);

    int n;
    n = LSEQ * DMODEL;
    f32_to_bf16_kernel<<<(n + 255) / 256, 256, 0, stream>>>(x, xbf, n);
    n = 2 * DINNER * DMODEL;
    f32_to_bf16_kernel<<<(n + 255) / 256, 256, 0, stream>>>(w_in, winbf, n);
    n = NXPAD * DINNER;
    pack_wxproj_kernel<<<(n + 255) / 256, 256, 0, stream>>>(w_xproj, wxbf);
    n = DMODEL * DINNER;
    f32_to_bf16_kernel<<<(n + 255) / 256, 256, 0, stream>>>(w_out, woutbf, n);
    n = DINNER * RPAD;
    pack_wdt_kernel<<<(n + 255) / 256, 256, 0, stream>>>(w_dt, wdtbf);

    dim3 gblk(GEMM_WAVES * 32);
    const int mRows = 16 * GEMM_MT * GEMM_WAVES;   // 128 rows of M per block

    // in_proj: xz[L, 3072] = xbf[L,768] x winbf[3072,768]^T
    gemm_bf16_wmma<<<dim3(2 * DINNER / 64, LSEQ / mRows), gblk, 0, stream>>>(
        xbf, winbf, xz, LSEQ, 2 * DINNER, DMODEL);

    // depthwise conv + SiLU
    n = LSEQ * DINNER;
    conv_silu_kernel<<<(n + 255) / 256, 256, 0, stream>>>(xz, w_conv, b_conv, xc, xcbf);

    // x_proj: x_dbl[L, NXPAD] = xcbf[L,1536] x wxbf[NXPAD,1536]^T (padded, branch-free)
    gemm_bf16_wmma<<<dim3(NXPAD / 64, LSEQ / mRows), gblk, 0, stream>>>(
        xcbf, wxbf, x_dbl, LSEQ, NXPAD, DINNER);

    // dt input pack (K padded 48 -> 64)
    n = LSEQ * RPAD;
    pack_dtin_kernel<<<(n + 255) / 256, 256, 0, stream>>>(x_dbl, dtinbf);

    // dt_proj: dt_raw[L, 1536] = dtinbf[L,64] x wdtbf[1536,64]^T
    gemm_bf16_wmma<<<dim3(DINNER / 64, LSEQ / mRows), gblk, 0, stream>>>(
        dtinbf, wdtbf, dt_raw, LSEQ, DINNER, RPAD);

    // selective scan (sequential over L, parallel over DINNER channels)
    scan_kernel<<<DINNER / SCAN_TPB, SCAN_TPB, 0, stream>>>(
        dt_raw, b_dt, x_dbl, xc, xz, A_log, D_param, ybf);

    // out_proj: outlin[L, 768] = ybf[L,1536] x woutbf[768,1536]^T
    gemm_bf16_wmma<<<dim3(DMODEL / 64, LSEQ / mRows), gblk, 0, stream>>>(
        ybf, woutbf, outlin, LSEQ, DMODEL, DINNER);

    // residual + RMSNorm
    rmsnorm_kernel<<<LSEQ, 256, 0, stream>>>(outlin, x, w_norm, (float*)d_out);
}
